// Descriptor_26001732010026
// MI455X (gfx1250) — compile-verified, matrix operand order checked by reading, not run on hardware
//
#include <hip/hip_runtime.h>
#include <hip/hip_bf16.h>
#include <math.h>

// ---------------------------------------------------------------------------
// AttentiveFP-style GNN forward for MI455X (gfx1250, wave32, WMMA).
// All Linear layers route through one bf16-in / f32-accumulate WMMA GEMM
// (v_wmma_f32_16x16x32_bf16), 128x64 block tile, 4 WMMAs per wave per K-step.
// K-parity and activation are template parameters so the hot loop carries no
// runtime branches; main K-loop is guard-free (M % 128 == 0; OOB B-rows only
// feed never-stored columns); only the ragged K-tail tile clamps/zero-fills.
// ---------------------------------------------------------------------------

typedef __attribute__((ext_vector_type(16))) __bf16 v16bf;
typedef __attribute__((ext_vector_type(8)))  float  v8f;

#define ACT_NONE  0
#define ACT_LRELU 1
#define ACT_RELU  2
#define ACT_ELU   3

#define NEG_MASK (-9e8f)

// Problem constants
#define Bn 128
#define Ln 256
#define Mn 8
#define Dn 200
#define FA 39
#define FB 10

// ---------------- WMMA GEMM:  C[M,N] = act( A[M,K] @ W[N,K]^T + bias*rs ) ---
// Requirement: M is a multiple of TILE_M (true for all call sites: 32768,
// 262144, 128). N and K are arbitrary.
#define TILE_M 128
#define TILE_N 64
#define TILE_K 32
#define LDSA_STRIDE 36   // bf16 elements, even (32-bit pack stores stay aligned)
#define LDSB_STRIDE 36

union FragBF { v16bf v; unsigned u[8]; };

__device__ __forceinline__ unsigned pack_bf16(float a, float b) {
#if __has_builtin(__builtin_amdgcn_cvt_pk_bf16_f32)
    typedef __attribute__((ext_vector_type(2))) __bf16 v2bf;
    union { v2bf h; unsigned u; } t;
    t.h = __builtin_amdgcn_cvt_pk_bf16_f32(a, b);
    return t.u;
#else
    union { __bf16 h[2]; unsigned u; } t;
    t.h[0] = (__bf16)a;
    t.h[1] = (__bf16)b;
    return t.u;
#endif
}

__device__ __forceinline__ void mma_step(const __bf16 (*As)[LDSA_STRIDE],
                                         const __bf16 (*Bs)[LDSB_STRIDE],
                                         int wave, int half, int mrow, v8f acc[4])
{
    // Fragments per ISA 7.12.2 16-bit layout:
    // lanes 0-15: K 0-7 (v0-3), 16-23 (v4-7); lanes 16-31: K 8-15, 24-31.
    const int kb = half * 8;
    FragBF fa;
    const int ar = wave * 16 + mrow;
    #pragma unroll
    for (int v = 0; v < 4; ++v) {
        fa.u[v]     = *(const unsigned*)&As[ar][kb + 2 * v];
        fa.u[4 + v] = *(const unsigned*)&As[ar][16 + kb + 2 * v];
    }
    #pragma unroll
    for (int s = 0; s < 4; ++s) {
        FragBF fb;
        const int br = s * 16 + mrow;
        #pragma unroll
        for (int v = 0; v < 4; ++v) {
            fb.u[v]     = *(const unsigned*)&Bs[br][kb + 2 * v];
            fb.u[4 + v] = *(const unsigned*)&Bs[br][16 + kb + 2 * v];
        }
        acc[s] = __builtin_amdgcn_wmma_f32_16x16x32_bf16(
            false, fa.v, false, fb.v, (short)0, acc[s], false, false);
    }
}

template<int ACT, bool EVENK>
__global__ __launch_bounds__(256)
void gemm_bias_act_wmma(const float* __restrict__ A, const float* __restrict__ W,
                        const float* __restrict__ bias, const float* __restrict__ rowscale,
                        float* __restrict__ C, int M, int N, int K)
{
    __shared__ __bf16 As[TILE_M][LDSA_STRIDE];
    __shared__ __bf16 Bs[TILE_N][LDSB_STRIDE];

    const int tid  = threadIdx.x;
    const int wave = tid >> 5;         // 8 waves (wave32)
    const int half = (tid >> 4) & 1;   // lane 0-15 vs 16-31
    const int mrow = tid & 15;
    const int m0 = blockIdx.x * TILE_M;   // always in range (M % TILE_M == 0)
    const int n0 = blockIdx.y * TILE_N;

    // Staging mapping: 16 threads per row-group; thread owns bf16 pair kS of
    // rows {rS, rS+16, ...}.
    const int rS = tid >> 4;           // 0..15
    const int kS = (tid & 15) * 2;     // 0,2,..,30

    // Per-thread base pointers (hoisted): A rows always valid; B rows clamped
    // (OOB rows feed only columns gc >= N, which are never stored).
    const float* pa = A + (size_t)(m0 + rS) * K + kS;
    const float* pb[4];
    #pragma unroll
    for (int i = 0; i < 4; ++i) {
        int gn = n0 + i * 16 + rS;
        pb[i] = W + (size_t)(gn < N ? gn : N - 1) * K + kS;
    }

    v8f acc[4] = {};
    const int kFull = K & ~(TILE_K - 1);

    for (int k0 = 0; k0 < kFull; k0 += TILE_K) {
        float ax[8], ay[8], bx[4], by[4];
        if constexpr (EVENK) {           // float2 loads stay 8B-aligned
            #pragma unroll
            for (int i = 0; i < 8; ++i) {
                float2 t = *(const float2*)(pa + (size_t)(i * 16) * K + k0);
                ax[i] = t.x; ay[i] = t.y;
            }
            #pragma unroll
            for (int i = 0; i < 4; ++i) {
                float2 t = *(const float2*)(pb[i] + k0);
                bx[i] = t.x; by[i] = t.y;
            }
        } else {
            #pragma unroll
            for (int i = 0; i < 8; ++i) {
                const float* p = pa + (size_t)(i * 16) * K + k0;
                ax[i] = p[0]; ay[i] = p[1];
            }
            #pragma unroll
            for (int i = 0; i < 4; ++i) {
                const float* p = pb[i] + k0;
                bx[i] = p[0]; by[i] = p[1];
            }
        }
        #pragma unroll
        for (int i = 0; i < 8; ++i)
            *(unsigned*)&As[i * 16 + rS][kS] = pack_bf16(ax[i], ay[i]);
        #pragma unroll
        for (int i = 0; i < 4; ++i)
            *(unsigned*)&Bs[i * 16 + rS][kS] = pack_bf16(bx[i], by[i]);
        __syncthreads();
        mma_step(As, Bs, wave, half, mrow, acc);
        __syncthreads();
    }

    if (kFull < K) {
        // Ragged K tail: clamp + zero-fill (runs at most once).
        const int gk0 = kFull + kS, gk1 = gk0 + 1;
        const int o0 = (gk0 < K ? gk0 : K - 1) - kS;   // offset from base ptrs
        const int o1 = (gk1 < K ? gk1 : K - 1) - kS;
        float ax[8], ay[8], bx[4], by[4];
        #pragma unroll
        for (int i = 0; i < 8; ++i) {
            const float* p = pa + (size_t)(i * 16) * K;
            ax[i] = p[o0]; ay[i] = p[o1];
        }
        #pragma unroll
        for (int i = 0; i < 4; ++i) {
            bx[i] = pb[i][o0]; by[i] = pb[i][o1];
        }
        #pragma unroll
        for (int i = 0; i < 8; ++i) {
            float v0 = gk0 < K ? ax[i] : 0.f;
            float v1 = gk1 < K ? ay[i] : 0.f;
            *(unsigned*)&As[i * 16 + rS][kS] = pack_bf16(v0, v1);
        }
        #pragma unroll
        for (int i = 0; i < 4; ++i) {
            float v0 = gk0 < K ? bx[i] : 0.f;
            float v1 = gk1 < K ? by[i] : 0.f;
            *(unsigned*)&Bs[i * 16 + rS][kS] = pack_bf16(v0, v1);
        }
        __syncthreads();
        mma_step(As, Bs, wave, half, mrow, acc);
        __syncthreads();
    }

    // Epilogue: VGPR r -> M = r (lanes 0-15) / r+8 (lanes 16-31); N = mrow.
    #pragma unroll
    for (int s = 0; s < 4; ++s) {
        int gc = n0 + s * 16 + mrow;
        if (gc < N) {
            float bscale = bias[gc];
            #pragma unroll
            for (int r = 0; r < 8; ++r) {
                int gr = m0 + wave * 16 + half * 8 + r;
                float v = acc[s][r];
                if constexpr (ACT == ACT_ELU)
                    v += bscale * rowscale[gr];   // folded softmax-weight sum
                else
                    v += bscale;
                if constexpr (ACT == ACT_LRELU)      v = v > 0.f ? v : 0.01f * v;
                else if constexpr (ACT == ACT_RELU)  v = fmaxf(v, 0.f);
                else if constexpr (ACT == ACT_ELU)   v = v > 0.f ? v : (__expf(v) - 1.f);
                C[(size_t)gr * N + gc] = v;
            }
        }
    }
}

// ---- gather concat(atom_nb, bond_nb) -> [B*L*M, 49] ------------------------
__global__ void k_gather_concat(const float* __restrict__ atom, const float* __restrict__ bond,
                                const int* __restrict__ adeg, const int* __restrict__ bdeg,
                                float* __restrict__ out, int total)
{
    int i = blockIdx.x * blockDim.x + threadIdx.x;
    if (i >= total) return;
    int f = i % (FA + FB);
    int r = i / (FA + FB);          // (b*L+l)*M+m
    int bl = r / Mn;
    int b  = bl / Ln;
    if (f < FA) {
        int idx = adeg[r];
        out[i] = atom[((size_t)b * Ln + idx) * FA + f];
    } else {
        int idx = bdeg[r];
        out[i] = bond[((size_t)b * Ln + idx) * FB + (f - FA)];
    }
}

// ---- gather 200-float rows: dst[b,l,m,:] = src[b, adeg[b,l,m], :] ----------
__global__ void k_gather_rows(const float* __restrict__ src, const int* __restrict__ adeg,
                              float* __restrict__ dst, int total)
{
    int i = blockIdx.x * blockDim.x + threadIdx.x;
    if (i >= total) return;
    int j = i % Dn;
    int r = i / Dn;                 // (b*L+l)*M+m
    int bl = r / Mn;
    int b  = bl / Ln;
    int idx = adeg[r];
    dst[i] = src[((size_t)b * Ln + idx) * Dn + j];
}

// ---- attention score per (b,l,m) -------------------------------------------
__global__ void k_score(const float* __restrict__ cur, const float* __restrict__ nbf,
                        const float* __restrict__ Wal, const float* __restrict__ bal,
                        const int* __restrict__ adeg, float* __restrict__ score, int total)
{
    int i = blockIdx.x * blockDim.x + threadIdx.x;
    if (i >= total) return;
    int bl = i / Mn;
    const float* c = cur + (size_t)bl * Dn;
    const float* n = nbf + (size_t)i * Dn;
    float s = bal[0];
    #pragma unroll 4
    for (int j = 0; j < Dn; ++j) s += Wal[j] * c[j];
    #pragma unroll 4
    for (int j = 0; j < Dn; ++j) s += Wal[Dn + j] * n[j];
    s = s > 0.f ? s : 0.01f * s;
    if (adeg[i] == Ln - 1) s += NEG_MASK;
    score[i] = s;
}

// ---- softmax over m + weighted neighbor sum --------------------------------
__global__ __launch_bounds__(256)
void k_softmax_ctx(const float* __restrict__ score, const float* __restrict__ nbf,
                   const int* __restrict__ adeg, float* __restrict__ wsum,
                   float* __restrict__ swout)
{
    int bl = blockIdx.x;
    __shared__ float s_w[Mn];
    if (threadIdx.x == 0) {
        float sc[Mn], mx = -1e30f;
        for (int m = 0; m < Mn; ++m) { sc[m] = score[bl * Mn + m]; mx = fmaxf(mx, sc[m]); }
        float e[Mn], sum = 0.f;
        for (int m = 0; m < Mn; ++m) { e[m] = __expf(sc[m] - mx); sum += e[m]; }
        float sw = 0.f;
        for (int m = 0; m < Mn; ++m) {
            float w = e[m] / sum;
            if (adeg[bl * Mn + m] == Ln - 1) w = 0.f;   // attend_mask
            s_w[m] = w; sw += w;
        }
        swout[bl] = sw;
    }
    __syncthreads();
    for (int j = threadIdx.x; j < Dn; j += blockDim.x) {
        float a = 0.f;
        #pragma unroll
        for (int m = 0; m < Mn; ++m)
            a += s_w[m] * nbf[((size_t)bl * Mn + m) * Dn + j];
        wsum[(size_t)bl * Dn + j] = a;
    }
}

// ---- GRU gate fusion: h <- GRU(gi, gh, h); act <- relu(h) ------------------
__global__ void k_gru_combine(const float* __restrict__ gi, const float* __restrict__ gh,
                              float* __restrict__ h, float* __restrict__ act, int rows)
{
    int i = blockIdx.x * blockDim.x + threadIdx.x;
    int total = rows * Dn;
    if (i >= total) return;
    int row = i / Dn, j = i % Dn;
    const float* gir = gi + (size_t)row * 3 * Dn;
    const float* ghr = gh + (size_t)row * 3 * Dn;
    float r = 1.f / (1.f + __expf(-(gir[j] + ghr[j])));
    float z = 1.f / (1.f + __expf(-(gir[Dn + j] + ghr[Dn + j])));
    float n = tanhf(gir[2 * Dn + j] + r * ghr[2 * Dn + j]);
    float hv = h[i];
    float o = (1.f - z) * n + z * hv;
    h[i] = o;
    act[i] = fmaxf(o, 0.f);
}

// ---- masked sum over L -> mol_feature, relu --------------------------------
__global__ void k_mol_pool(const float* __restrict__ act, const float* __restrict__ am,
                           float* __restrict__ molf, float* __restrict__ actmol)
{
    int b = blockIdx.x, j = threadIdx.x;
    if (j >= Dn) return;
    float s = 0.f;
    for (int l = 0; l < Ln; ++l)
        s += act[((size_t)b * Ln + l) * Dn + j] * am[b * Ln + l];
    molf[b * Dn + j] = s;
    actmol[b * Dn + j] = fmaxf(s, 0.f);
}

// ---- mol attention score per (b,l) -----------------------------------------
__global__ void k_mol_score(const float* __restrict__ actmol, const float* __restrict__ act,
                            const float* __restrict__ Wm, const float* __restrict__ bm,
                            const float* __restrict__ am, float* __restrict__ score)
{
    int i = blockIdx.x * blockDim.x + threadIdx.x;
    if (i >= Bn * Ln) return;
    int b = i / Ln;
    const float* x = actmol + (size_t)b * Dn;
    const float* y = act + (size_t)i * Dn;
    float s = bm[0];
    #pragma unroll 4
    for (int j = 0; j < Dn; ++j) s += Wm[j] * x[j];
    #pragma unroll 4
    for (int j = 0; j < Dn; ++j) s += Wm[Dn + j] * y[j];
    s = s > 0.f ? s : 0.01f * s;
    if (am[i] == 0.f) s += NEG_MASK;
    score[i] = s;
}

// ---- mol softmax over L + weighted sum -------------------------------------
__global__ __launch_bounds__(256)
void k_mol_softmax_ctx(const float* __restrict__ score, const float* __restrict__ act,
                       const float* __restrict__ am, float* __restrict__ mwsum,
                       float* __restrict__ msw)
{
    int b = blockIdx.x, t = threadIdx.x;
    __shared__ float s_w[Ln];
    __shared__ float red[Ln];
    float sc = score[b * Ln + t];
    red[t] = sc; __syncthreads();
    for (int s = Ln / 2; s > 0; s >>= 1) { if (t < s) red[t] = fmaxf(red[t], red[t + s]); __syncthreads(); }
    float mx = red[0]; __syncthreads();
    float e = __expf(sc - mx);
    red[t] = e; __syncthreads();
    for (int s = Ln / 2; s > 0; s >>= 1) { if (t < s) red[t] += red[t + s]; __syncthreads(); }
    float sum = red[0]; __syncthreads();
    float w = (e / sum) * am[b * Ln + t];
    s_w[t] = w;
    red[t] = w; __syncthreads();
    for (int s = Ln / 2; s > 0; s >>= 1) { if (t < s) red[t] += red[t + s]; __syncthreads(); }
    if (t == 0) msw[b] = red[0];
    __syncthreads();
    for (int j = t; j < Dn; j += Ln) {
        float a = 0.f;
        for (int l = 0; l < Ln; ++l)
            a += s_w[l] * act[((size_t)b * Ln + l) * Dn + j];
        mwsum[b * Dn + j] = a;
    }
}

// ---- concat [rows,na]+[rows,nb] --------------------------------------------
__global__ void k_concat2(const float* __restrict__ a, const float* __restrict__ b,
                          float* __restrict__ out, int na, int nb, int rows)
{
    int i = blockIdx.x * blockDim.x + threadIdx.x;
    int n = na + nb, total = rows * n;
    if (i >= total) return;
    int r = i / n, c = i % n;
    out[i] = (c < na) ? a[r * na + c] : b[r * nb + (c - na)];
}

// ---------------------------------------------------------------------------
extern "C" void kernel_launch(void* const* d_in, const int* in_sizes, int n_in,
                              void* d_out, int out_size, void* d_ws, size_t ws_size,
                              hipStream_t stream)
{
    (void)in_sizes; (void)n_in; (void)out_size; (void)ws_size;

    // Inputs in setup_inputs() insertion order; params dict expanded in place.
    const float* atom_list   = (const float*)d_in[0];   // [128,256,39]
    const float* bond_list   = (const float*)d_in[1];   // [128,256,10]
    const float* atom_mask   = (const float*)d_in[2];   // [128,256]
    const float* descriptors = (const float*)d_in[3];   // [128,200]
    const float* Wa        = (const float*)d_in[4];
    const float* ba        = (const float*)d_in[5];
    const float* Wn        = (const float*)d_in[6];
    const float* bn        = (const float*)d_in[7];
    const float* gru_wih   = (const float*)d_in[8];     // [3,600,200]
    const float* gru_whh   = (const float*)d_in[9];
    const float* gru_bih   = (const float*)d_in[10];    // [3,600]
    const float* gru_bhh   = (const float*)d_in[11];
    const float* W_align   = (const float*)d_in[12];    // [3,1,400]
    const float* b_align   = (const float*)d_in[13];    // [3,1]
    const float* W_attend  = (const float*)d_in[14];    // [3,200,200]
    const float* b_attend  = (const float*)d_in[15];    // [3,200]
    const float* mg_wih    = (const float*)d_in[16];
    const float* mg_whh    = (const float*)d_in[17];
    const float* mg_bih    = (const float*)d_in[18];
    const float* mg_bhh    = (const float*)d_in[19];
    const float* W_malign  = (const float*)d_in[20];
    const float* b_malign  = (const float*)d_in[21];
    const float* W_mattend = (const float*)d_in[22];
    const float* b_mattend = (const float*)d_in[23];
    const float* W_fcg1    = (const float*)d_in[24];
    const float* b_fcg1    = (const float*)d_in[25];
    const float* W_sn1     = (const float*)d_in[26];
    const float* b_sn1     = (const float*)d_in[27];
    const float* W_sn2     = (const float*)d_in[28];
    const float* b_sn2     = (const float*)d_in[29];
    const float* W_sn3     = (const float*)d_in[30];
    const float* b_sn3     = (const float*)d_in[31];
    const float* W_fc1     = (const float*)d_in[32];
    const float* b_fc1     = (const float*)d_in[33];
    const float* W_fc2     = (const float*)d_in[34];
    const float* b_fc2     = (const float*)d_in[35];
    const float* W_out     = (const float*)d_in[36];
    const float* b_out     = (const float*)d_in[37];
    const int* adeg = (const int*)d_in[38];             // [128,256,8]
    const int* bdeg = (const int*)d_in[39];

    const int BL  = Bn * Ln;          // 32768
    const int BLM = BL * Mn;          // 262144

    // Workspace carve-out
    char* ws = (char*)d_ws;
    size_t off = 0;
    auto carve = [&](size_t bytes) -> float* {
        float* p = (float*)(ws + off);
        off = (off + bytes + 255) & ~(size_t)255;
        return p;
    };
    float* h     = carve((size_t)BL * Dn * 4);     // atom_feature / hidden
    float* act   = carve((size_t)BL * Dn * 4);     // relu(h)
    float* nbin  = carve((size_t)BLM * (FA + FB) * 4);
    float* nbf   = carve((size_t)BLM * Dn * 4);    // neighbor features
    float* score = carve((size_t)BLM * 4);
    float* wsum  = carve((size_t)BL * Dn * 4);
    float* sw    = carve((size_t)BL * 4);
    float* ctx   = carve((size_t)BL * Dn * 4);
    float* gi    = carve((size_t)BL * 3 * Dn * 4);
    float* gh    = carve((size_t)BL * 3 * Dn * 4);
    float* molf   = carve((size_t)Bn * Dn * 4);
    float* actmol = carve((size_t)Bn * Dn * 4);
    float* mscore = carve((size_t)Bn * Ln * 4);
    float* mwsum  = carve((size_t)Bn * Dn * 4);
    float* msw    = carve((size_t)Bn * 4);
    float* mctx   = carve((size_t)Bn * Dn * 4);
    float* mgi    = carve((size_t)Bn * 3 * Dn * 4);
    float* mgh    = carve((size_t)Bn * 3 * Dn * 4);
    float* xbuf   = carve((size_t)Bn * 128 * 4);
    float* dsc1   = carve((size_t)Bn * 512 * 4);
    float* dsc2   = carve((size_t)Bn * 1024 * 4);
    float* dsc3   = carve((size_t)Bn * Dn * 4);
    float* cat    = carve((size_t)Bn * 328 * 4);
    float* m1     = carve((size_t)Bn * 1312 * 4);
    float* m2     = carve((size_t)Bn * 656 * 4);

    // Dispatch: only 4 instantiations occur.
    //   odd K  (39, 49)            -> LRELU
    //   even K (200/512/1024/...)  -> NONE / RELU / ELU(+rowscale)
    auto gemm = [&](const float* A, const float* W, const float* bias,
                    const float* rscale, float* C, int M, int N, int K, int a) {
        dim3 g(M / TILE_M, (N + TILE_N - 1) / TILE_N);
        if (K & 1) {
            gemm_bias_act_wmma<ACT_LRELU, false><<<g, 256, 0, stream>>>(A, W, bias, rscale, C, M, N, K);
        } else if (a == ACT_ELU) {
            gemm_bias_act_wmma<ACT_ELU, true><<<g, 256, 0, stream>>>(A, W, bias, rscale, C, M, N, K);
        } else if (a == ACT_RELU) {
            gemm_bias_act_wmma<ACT_RELU, true><<<g, 256, 0, stream>>>(A, W, bias, rscale, C, M, N, K);
        } else {
            gemm_bias_act_wmma<ACT_NONE, true><<<g, 256, 0, stream>>>(A, W, bias, rscale, C, M, N, K);
        }
    };
    auto blocks = [](int total) { return (total + 255) / 256; };

    // atom_feature = lrelu(atom_list @ Wa^T + ba)  -> h
    gemm(atom_list, Wa, ba, nullptr, h, BL, Dn, FA, ACT_LRELU);

    // neighbor_feature = lrelu(concat(gather) @ Wn^T + bn) -> nbf
    int tot_nbin = BLM * (FA + FB);
    k_gather_concat<<<blocks(tot_nbin), 256, 0, stream>>>(atom_list, bond_list, adeg, bdeg, nbin, tot_nbin);
    gemm(nbin, Wn, bn, nullptr, nbf, BLM, Dn, FA + FB, ACT_LRELU);

    // ---- radius loop ----
    for (int d = 0; d < 3; ++d) {
        const float* cur = (d == 0) ? h : act;
        if (d > 0) {
            int tot = BLM * Dn;
            k_gather_rows<<<blocks(tot), 256, 0, stream>>>(act, adeg, nbf, tot);
        }
        k_score<<<blocks(BLM), 256, 0, stream>>>(cur, nbf, W_align + d * 2 * Dn,
                                                 b_align + d, adeg, score, BLM);
        k_softmax_ctx<<<BL, 256, 0, stream>>>(score, nbf, adeg, wsum, sw);
        // context = elu(wsum @ W_attend^T + sw * b_attend)
        gemm(wsum, W_attend + (size_t)d * Dn * Dn, b_attend + d * Dn, sw, ctx, BL, Dn, Dn, ACT_ELU);
        // GRU gates
        gemm(ctx, gru_wih + (size_t)d * 3 * Dn * Dn, gru_bih + d * 3 * Dn, nullptr, gi, BL, 3 * Dn, Dn, ACT_NONE);
        gemm(h,   gru_whh + (size_t)d * 3 * Dn * Dn, gru_bhh + d * 3 * Dn, nullptr, gh, BL, 3 * Dn, Dn, ACT_NONE);
        k_gru_combine<<<blocks(BL * Dn), 256, 0, stream>>>(gi, gh, h, act, BL);
    }

    // ---- molecule readout ----
    k_mol_pool<<<Bn, 256, 0, stream>>>(act, atom_mask, molf, actmol);
    for (int t = 0; t < 2; ++t) {
        k_mol_score<<<blocks(Bn * Ln), 256, 0, stream>>>(actmol, act, W_malign, b_malign, atom_mask, mscore);
        k_mol_softmax_ctx<<<Bn, Ln, 0, stream>>>(mscore, act, atom_mask, mwsum, msw);
        gemm(mwsum, W_mattend, b_mattend, msw, mctx, Bn, Dn, Dn, ACT_ELU);
        gemm(mctx, mg_wih, mg_bih, nullptr, mgi, Bn, 3 * Dn, Dn, ACT_NONE);
        gemm(molf, mg_whh, mg_bhh, nullptr, mgh, Bn, 3 * Dn, Dn, ACT_NONE);
        k_gru_combine<<<blocks(Bn * Dn), 256, 0, stream>>>(mgi, mgh, molf, actmol, Bn);
    }

    // ---- head ----
    gemm(molf, W_fcg1, b_fcg1, nullptr, xbuf, Bn, 128, Dn, ACT_NONE);
    gemm(descriptors, W_sn1, b_sn1, nullptr, dsc1, Bn, 512, Dn, ACT_RELU);
    gemm(dsc1, W_sn2, b_sn2, nullptr, dsc2, Bn, 1024, 512, ACT_RELU);
    gemm(dsc2, W_sn3, b_sn3, nullptr, dsc3, Bn, Dn, 1024, ACT_NONE);
    k_concat2<<<blocks(Bn * 328), 256, 0, stream>>>(xbuf, dsc3, cat, 128, Dn, Bn);
    gemm(cat, W_fc1, b_fc1, nullptr, m1, Bn, 1312, 328, ACT_RELU);
    gemm(m1, W_fc2, b_fc2, nullptr, m2, Bn, 656, 1312, ACT_RELU);

    // Outputs: atom_feature [B,L,D] then pred [B,1], concatenated flat (f32).
    float* out_atom = (float*)d_out;
    float* out_pred = out_atom + (size_t)BL * Dn;
    gemm(m2, W_out, b_out, nullptr, out_pred, Bn, 1, 656, ACT_NONE);
    hipMemcpyAsync(out_atom, h, (size_t)BL * Dn * sizeof(float),
                   hipMemcpyDeviceToDevice, stream);
}
